// Attention_17085379903530
// MI455X (gfx1250) — matrix operand order repaired and checked by reading, not compiled, not run on hardware
//
#include <hip/hip_runtime.h>

typedef __attribute__((ext_vector_type(16))) _Float16 v16h;
typedef __attribute__((ext_vector_type(8)))  _Float16 v8h;
typedef __attribute__((ext_vector_type(4)))  _Float16 v4h;
typedef __attribute__((ext_vector_type(8)))  float    v8f;
typedef __attribute__((ext_vector_type(4)))  float    v4f;

#define S_LEN  4096
#define DMODEL 1024
#define NHEAD  16
#define HDIM   64
#define SM_SCALE 0.125f   // 1/sqrt(64)

// ---------------------------------------------------------------- WMMA helpers

__device__ __forceinline__ v8f wmma16x16x32(v16h a, v16h b, v8f c) {
  // D = A(16x32 f16) * B(32x16 f16) + C(16x16 f32)
  return __builtin_amdgcn_wmma_f32_16x16x32_f16(false, a, false, b, (short)0, c,
                                                false, false);
}

// A fragment 16x32 (MxK), A row-major with leading dim lda.
// Lane L<16 : row M=L, K {k0..k0+7} in v[0..3], K {k0+16..k0+23} in v[4..7]
// Lane L>=16: row M=L-16, K {k0+8..k0+15}, {k0+24..k0+31}
__device__ __forceinline__ v16h load_A(const _Float16* A, int lda, int row0,
                                       int k0, int lane) {
  int m    = row0 + (lane & 15);
  int koff = (lane & 16) ? 8 : 0;
  const _Float16* p = A + m * lda + k0 + koff;
  v16h a;
  *reinterpret_cast<v8h*>(&a)       = *reinterpret_cast<const v8h*>(p);
  *(reinterpret_cast<v8h*>(&a) + 1) = *reinterpret_cast<const v8h*>(p + 16);
  return a;
}

// B fragment 32x16 (KxN) where B[k][n] = W[n][k], W row-major (ldw).
// Lane L<16 : col N=L, K {k0..k0+7}, {k0+16..k0+23}; Lane L>=16: col N=L-16, K {k0+8..k0+15}, {k0+24..k0+31}
__device__ __forceinline__ v16h load_Bt(const _Float16* W, int ldw, int n0,
                                        int k0, int lane) {
  int n  = n0 + (lane & 15);
  int kb = k0 + ((lane & 16) ? 8 : 0);
  const _Float16* p = W + n * ldw + kb;
  v16h b;
  *reinterpret_cast<v8h*>(&b)       = *reinterpret_cast<const v8h*>(p);
  *(reinterpret_cast<v8h*>(&b) + 1) = *reinterpret_cast<const v8h*>(p + 16);
  return b;
}

// ---------------------------------------------------------------- fp32 -> fp16

__global__ void __launch_bounds__(256)
cvt_f32_f16(const float* __restrict__ src, _Float16* __restrict__ dst, int n4) {
  int i = blockIdx.x * blockDim.x + threadIdx.x;
  if (i < n4) {
    v4f v = reinterpret_cast<const v4f*>(src)[i];
    v4h h;
    h.x = (_Float16)v.x; h.y = (_Float16)v.y;
    h.z = (_Float16)v.z; h.w = (_Float16)v.w;
    reinterpret_cast<v4h*>(dst)[i] = h;
  }
}

// ---------------------------------------------------------------- QKV GEMM
// x[4096,1024] @ {wq|wk|wv}^T -> Q,K as [H, S, 64]; V transposed as [H, 64, S]
// Wave tile: 32 rows x 64 cols, software-pipelined (double-buffered frags).

__global__ void __launch_bounds__(256)
qkv_gemm(const _Float16* __restrict__ x, const _Float16* __restrict__ w3,
         _Float16* __restrict__ qb, _Float16* __restrict__ kb,
         _Float16* __restrict__ vt) {
  int lane = threadIdx.x & 31;
  int wid  = threadIdx.x >> 5;
  int m0 = blockIdx.y * 256 + wid * 32;   // 32 query rows per wave
  int n0 = blockIdx.x * 64;               // 64 output cols (over 3*1024)

  v8f acc0[4] = {}, acc1[4] = {};

  v16h a0 = load_A(x, DMODEL, m0,      0, lane);
  v16h a1 = load_A(x, DMODEL, m0 + 16, 0, lane);
  v16h b0 = load_Bt(w3, DMODEL, n0,      0, lane);
  v16h b1 = load_Bt(w3, DMODEL, n0 + 16, 0, lane);
  v16h b2 = load_Bt(w3, DMODEL, n0 + 32, 0, lane);
  v16h b3 = load_Bt(w3, DMODEL, n0 + 48, 0, lane);

  for (int k = 0; k < DMODEL - 32; k += 32) {
    int kn = k + 32;
    // issue next-step loads first so they overlap the 8 WMMAs below
    v16h na0 = load_A(x, DMODEL, m0,      kn, lane);
    v16h na1 = load_A(x, DMODEL, m0 + 16, kn, lane);
    v16h nb0 = load_Bt(w3, DMODEL, n0,      kn, lane);
    v16h nb1 = load_Bt(w3, DMODEL, n0 + 16, kn, lane);
    v16h nb2 = load_Bt(w3, DMODEL, n0 + 32, kn, lane);
    v16h nb3 = load_Bt(w3, DMODEL, n0 + 48, kn, lane);
    __builtin_prefetch(x + m0 * DMODEL + kn + 32, 0, 1);

    acc0[0] = wmma16x16x32(a0, b0, acc0[0]);
    acc0[1] = wmma16x16x32(a0, b1, acc0[1]);
    acc0[2] = wmma16x16x32(a0, b2, acc0[2]);
    acc0[3] = wmma16x16x32(a0, b3, acc0[3]);
    acc1[0] = wmma16x16x32(a1, b0, acc1[0]);
    acc1[1] = wmma16x16x32(a1, b1, acc1[1]);
    acc1[2] = wmma16x16x32(a1, b2, acc1[2]);
    acc1[3] = wmma16x16x32(a1, b3, acc1[3]);

    a0 = na0; a1 = na1;
    b0 = nb0; b1 = nb1; b2 = nb2; b3 = nb3;
  }
  // final k-step
  acc0[0] = wmma16x16x32(a0, b0, acc0[0]);
  acc0[1] = wmma16x16x32(a0, b1, acc0[1]);
  acc0[2] = wmma16x16x32(a0, b2, acc0[2]);
  acc0[3] = wmma16x16x32(a0, b3, acc0[3]);
  acc1[0] = wmma16x16x32(a1, b0, acc1[0]);
  acc1[1] = wmma16x16x32(a1, b1, acc1[1]);
  acc1[2] = wmma16x16x32(a1, b2, acc1[2]);
  acc1[3] = wmma16x16x32(a1, b3, acc1[3]);

  int widx   = n0 >> 10;          // 0=Q 1=K 2=V (uniform per block)
  int nin    = n0 & 1023;
  int h      = nin >> 6;          // head (n0 is a multiple of 64)
  int half8  = (lane & 16) ? 8 : 0;
  int lanelo = lane & 15;

  if (widx < 2) {
    _Float16* dst = (widx == 0 ? qb : kb) + h * (S_LEN * HDIM);
#pragma unroll
    for (int nt = 0; nt < 4; ++nt) {
      int d = nt * 16 + lanelo;
#pragma unroll
      for (int r = 0; r < 8; ++r) {
        int ma = m0 + r + half8;
        dst[ma * HDIM + d]        = (_Float16)acc0[nt][r];
        dst[(ma + 16) * HDIM + d] = (_Float16)acc1[nt][r];
      }
    }
  } else {
    _Float16* dst = vt + h * (HDIM * S_LEN);      // [d][s] transposed
#pragma unroll
    for (int nt = 0; nt < 4; ++nt) {
      int d = nt * 16 + lanelo;
#pragma unroll
      for (int r = 0; r < 8; ++r) {
        int ma = m0 + r + half8;
        dst[d * S_LEN + ma]      = (_Float16)acc0[nt][r];
        dst[d * S_LEN + ma + 16] = (_Float16)acc1[nt][r];
      }
    }
  }
}

// ---------------------------------------------------------------- RoPE (Q,K)

__global__ void __launch_bounds__(256)
rope_kernel(_Float16* __restrict__ qb, _Float16* __restrict__ kb,
            const float* __restrict__ cosb, const float* __restrict__ sinb) {
  int i = blockIdx.x * blockDim.x + threadIdx.x;   // NHEAD*S_LEN*32 pairs
  if (i >= NHEAD * S_LEN * (HDIM / 2)) return;
  int j = i & 31;
  int s = (i >> 5) & (S_LEN - 1);
  int h = i >> 17;
  float c  = cosb[s * 32 + j];
  float sn = sinb[s * 32 + j];
  int base = (h * S_LEN + s) * HDIM + 2 * j;
  float qe = (float)qb[base], qo = (float)qb[base + 1];
  qb[base]     = (_Float16)(qe * c - qo * sn);
  qb[base + 1] = (_Float16)(qe * sn + qo * c);
  float ke = (float)kb[base], ko = (float)kb[base + 1];
  kb[base]     = (_Float16)(ke * c - ko * sn);
  kb[base + 1] = (_Float16)(ke * sn + ko * c);
}

// ---------------------------------------------------------------- attention
// One wave per (head, 16-query tile). Flash-style online softmax over 32-key
// chunks; P bounced through a wave-private LDS tile to convert C-layout ->
// A-fragment layout for the P@V WMMAs. V fragments are fetched early so their
// global latency hides under the softmax VALU work.

__global__ void __launch_bounds__(256)
attn_kernel(const _Float16* __restrict__ qb, const _Float16* __restrict__ kb,
            const _Float16* __restrict__ vt, _Float16* __restrict__ ao) {
  __shared__ __align__(16) _Float16 ldsP[8][16 * 32];

  int lane   = threadIdx.x & 31;
  int wid    = threadIdx.x >> 5;
  int gw     = blockIdx.x * 8 + wid;
  int h      = gw >> 8;                 // 16 heads
  int qt     = gw & 255;                // 256 query tiles
  int m0     = qt * 16;
  int lanelo = lane & 15;
  int half8  = (lane & 16) ? 8 : 0;

  const _Float16* qh = qb + h * (S_LEN * HDIM);
  const _Float16* kh = kb + h * (S_LEN * HDIM);
  const _Float16* vh = vt + h * (HDIM * S_LEN);
  _Float16* lp = &ldsP[wid][0];

  v16h qA0 = load_A(qh, HDIM, m0, 0, lane);
  v16h qA1 = load_A(qh, HDIM, m0, 32, lane);

  v8f O0 = {}, O1 = {}, O2 = {}, O3 = {};
  float mi[8], li[8];
#pragma unroll
  for (int r = 0; r < 8; ++r) { mi[r] = -1.0e30f; li[r] = 0.0f; }

  int nch = (m0 + 47) >> 5;             // causal: keys 0 .. m0+15
  for (int ch = 0; ch < nch; ++ch) {
    int key0 = ch << 5;

    // all score B-fragments up front: one wait instead of four
    v16h kb00 = load_Bt(kh, HDIM, key0,      0,  lane);
    v16h kb01 = load_Bt(kh, HDIM, key0,      32, lane);
    v16h kb10 = load_Bt(kh, HDIM, key0 + 16, 0,  lane);
    v16h kb11 = load_Bt(kh, HDIM, key0 + 16, 32, lane);
    // V fragments issued early; consumed only after the softmax block
    v16h vb0 = load_Bt(vh, S_LEN, 0,  key0, lane);
    v16h vb1 = load_Bt(vh, S_LEN, 16, key0, lane);
    v16h vb2 = load_Bt(vh, S_LEN, 32, key0, lane);
    v16h vb3 = load_Bt(vh, S_LEN, 48, key0, lane);

    v8f s0 = {}, s1 = {};
    s0 = wmma16x16x32(qA0, kb00, s0);
    s0 = wmma16x16x32(qA1, kb01, s0);
    s1 = wmma16x16x32(qA0, kb10, s1);
    s1 = wmma16x16x32(qA1, kb11, s1);

    float alpha[8];
#pragma unroll
    for (int r = 0; r < 8; ++r) {
      int row = m0 + r + half8;
      int c0  = key0 + lanelo;
      float a0 = s0[r] * SM_SCALE;
      float a1 = s1[r] * SM_SCALE;
      if (c0 > row)      a0 = -1.0e30f;   // causal mask
      if (c0 + 16 > row) a1 = -1.0e30f;
      // row max across the 16 lanes of this half (row lives in one half)
      float t = fmaxf(a0, a1);
      t = fmaxf(t, __shfl_xor(t, 1, 32));
      t = fmaxf(t, __shfl_xor(t, 2, 32));
      t = fmaxf(t, __shfl_xor(t, 4, 32));
      t = fmaxf(t, __shfl_xor(t, 8, 32));
      float mn = fmaxf(mi[r], t);
      float p0 = __expf(a0 - mn);
      float p1 = __expf(a1 - mn);
      float rs = p0 + p1;
      rs += __shfl_xor(rs, 1, 32);
      rs += __shfl_xor(rs, 2, 32);
      rs += __shfl_xor(rs, 4, 32);
      rs += __shfl_xor(rs, 8, 32);
      float al = __expf(mi[r] - mn);
      mi[r] = mn;
      li[r] = li[r] * al + rs;
      alpha[r] = al;
      int lrow = (r + half8) * 32;
      lp[lrow + lanelo]      = (_Float16)p0;   // stage P row-major 16x32
      lp[lrow + 16 + lanelo] = (_Float16)p1;
    }
#pragma unroll
    for (int r = 0; r < 8; ++r) {
      O0[r] *= alpha[r]; O1[r] *= alpha[r];
      O2[r] *= alpha[r]; O3[r] *= alpha[r];
    }

    asm volatile("s_wait_dscnt 0x0" ::: "memory");  // P stores -> A-frag loads
    v16h pA = load_A(lp, 32, 0, 0, lane);
    O0 = wmma16x16x32(pA, vb0, O0);
    O1 = wmma16x16x32(pA, vb1, O1);
    O2 = wmma16x16x32(pA, vb2, O2);
    O3 = wmma16x16x32(pA, vb3, O3);
  }

#pragma unroll
  for (int r = 0; r < 8; ++r) {
    float inv = 1.0f / li[r];
    O0[r] *= inv; O1[r] *= inv; O2[r] *= inv; O3[r] *= inv;
  }
  int colbase = h * HDIM;
#pragma unroll
  for (int r = 0; r < 8; ++r) {
    int m = m0 + r + half8;
    _Float16* dst = ao + m * DMODEL + colbase;
    dst[lanelo]      = (_Float16)O0[r];
    dst[16 + lanelo] = (_Float16)O1[r];
    dst[32 + lanelo] = (_Float16)O2[r];
    dst[48 + lanelo] = (_Float16)O3[r];
  }
}

// ---------------------------------------------------------------- out proj
// ao[4096,1024] @ wo^T -> out (fp32). Same pipelined 32x64 wave tile.

__global__ void __launch_bounds__(256)
out_gemm(const _Float16* __restrict__ ao, const _Float16* __restrict__ wo,
         float* __restrict__ out) {
  int lane = threadIdx.x & 31;
  int wid  = threadIdx.x >> 5;
  int m0 = blockIdx.y * 256 + wid * 32;
  int n0 = blockIdx.x * 64;

  v8f acc0[4] = {}, acc1[4] = {};

  v16h a0 = load_A(ao, DMODEL, m0,      0, lane);
  v16h a1 = load_A(ao, DMODEL, m0 + 16, 0, lane);
  v16h b0 = load_Bt(wo, DMODEL, n0,      0, lane);
  v16h b1 = load_Bt(wo, DMODEL, n0 + 16, 0, lane);
  v16h b2 = load_Bt(wo, DMODEL, n0 + 32, 0, lane);
  v16h b3 = load_Bt(wo, DMODEL, n0 + 48, 0, lane);

  for (int k = 0; k < DMODEL - 32; k += 32) {
    int kn = k + 32;
    v16h na0 = load_A(ao, DMODEL, m0,      kn, lane);
    v16h na1 = load_A(ao, DMODEL, m0 + 16, kn, lane);
    v16h nb0 = load_Bt(wo, DMODEL, n0,      kn, lane);
    v16h nb1 = load_Bt(wo, DMODEL, n0 + 16, kn, lane);
    v16h nb2 = load_Bt(wo, DMODEL, n0 + 32, kn, lane);
    v16h nb3 = load_Bt(wo, DMODEL, n0 + 48, kn, lane);

    acc0[0] = wmma16x16x32(a0, b0, acc0[0]);
    acc0[1] = wmma16x16x32(a0, b1, acc0[1]);
    acc0[2] = wmma16x16x32(a0, b2, acc0[2]);
    acc0[3] = wmma16x16x32(a0, b3, acc0[3]);
    acc1[0] = wmma16x16x32(a1, b0, acc1[0]);
    acc1[1] = wmma16x16x32(a1, b1, acc1[1]);
    acc1[2] = wmma16x16x32(a1, b2, acc1[2]);
    acc1[3] = wmma16x16x32(a1, b3, acc1[3]);

    a0 = na0; a1 = na1;
    b0 = nb0; b1 = nb1; b2 = nb2; b3 = nb3;
  }
  acc0[0] = wmma16x16x32(a0, b0, acc0[0]);
  acc0[1] = wmma16x16x32(a0, b1, acc0[1]);
  acc0[2] = wmma16x16x32(a0, b2, acc0[2]);
  acc0[3] = wmma16x16x32(a0, b3, acc0[3]);
  acc1[0] = wmma16x16x32(a1, b0, acc1[0]);
  acc1[1] = wmma16x16x32(a1, b1, acc1[1]);
  acc1[2] = wmma16x16x32(a1, b2, acc1[2]);
  acc1[3] = wmma16x16x32(a1, b3, acc1[3]);

  int half8  = (lane & 16) ? 8 : 0;
  int lanelo = lane & 15;
#pragma unroll
  for (int nt = 0; nt < 4; ++nt) {
    int n = n0 + nt * 16 + lanelo;
#pragma unroll
    for (int r = 0; r < 8; ++r) {
      int m = m0 + r + half8;
      out[m * DMODEL + n]        = acc0[nt][r];
      out[(m + 16) * DMODEL + n] = acc1[nt][r];
    }
  }
}

// ---------------------------------------------------------------- launcher

extern "C" void kernel_launch(void* const* d_in, const int* in_sizes, int n_in,
                              void* d_out, int out_size, void* d_ws,
                              size_t ws_size, hipStream_t stream) {
  const float* x    = (const float*)d_in[0];
  const float* cosb = (const float*)d_in[1];
  const float* sinb = (const float*)d_in[2];
  // d_in[3] = mask (causal; implemented analytically), d_in[8] = start_pos (0)
  const float* wq = (const float*)d_in[4];
  const float* wk = (const float*)d_in[5];
  const float* wv = (const float*)d_in[6];
  const float* wo = (const float*)d_in[7];
  float* out = (float*)d_out;

  const size_t SD = (size_t)S_LEN * DMODEL;   // 4 Mi elements
  const size_t WW = (size_t)DMODEL * DMODEL;  // 1 Mi elements
  _Float16* x16 = (_Float16*)d_ws;            // workspace: 48 MB total
  _Float16* w16 = x16 + SD;                   // wq|wk|wv|wo stacked
  _Float16* qb  = w16 + 4 * WW;               // [H, S, 64]
  _Float16* kbf = qb + SD;                    // [H, S, 64]
  _Float16* vt  = kbf + SD;                   // [H, 64, S] (transposed V)
  _Float16* ao  = vt + SD;                    // [S, 1024] attention output

  cvt_f32_f16<<<(int)(SD / 4 / 256), 256, 0, stream>>>(x, x16, (int)(SD / 4));
  cvt_f32_f16<<<(int)(WW / 4 / 256), 256, 0, stream>>>(wq, w16 + 0 * WW, (int)(WW / 4));
  cvt_f32_f16<<<(int)(WW / 4 / 256), 256, 0, stream>>>(wk, w16 + 1 * WW, (int)(WW / 4));
  cvt_f32_f16<<<(int)(WW / 4 / 256), 256, 0, stream>>>(wv, w16 + 2 * WW, (int)(WW / 4));
  cvt_f32_f16<<<(int)(WW / 4 / 256), 256, 0, stream>>>(wo, w16 + 3 * WW, (int)(WW / 4));

  qkv_gemm<<<dim3(48, 16), 256, 0, stream>>>(x16, w16, qb, kbf, vt);
  rope_kernel<<<(NHEAD * S_LEN * (HDIM / 2)) / 256, 256, 0, stream>>>(qb, kbf,
                                                                      cosb, sinb);
  attn_kernel<<<(NHEAD * 256) / 8, 256, 0, stream>>>(qb, kbf, vt, ao);
  out_gemm<<<dim3(16, 16), 256, 0, stream>>>(ao, w16 + 3 * WW, out);
}
